// SpatioTemporalAttention_66460323938911
// MI455X (gfx1250) — compile-verified, matrix operand order checked by reading, not run on hardware
//
#include <hip/hip_runtime.h>
#include <hip/hip_bf16.h>
#include <stdint.h>

// ---------------------------------------------------------------------------
// Problem constants (from the reference)
// ---------------------------------------------------------------------------
#define BB 128
#define SS 120
#define JJ 24
#define EE 64
#define HH 4
#define AA 16          // attention dim = E/H
#define CC 1536        // J*E
#define NROW (BB * SS) // 15360 rows for the big GEMMs
#define EPSV 1e-5f

typedef __attribute__((ext_vector_type(16))) __bf16 v16bf;
typedef __attribute__((ext_vector_type(8)))  float  v8f;
typedef __attribute__((ext_vector_type(4)))  int    ivec4;
typedef __attribute__((ext_vector_type(8)))  int    ivec8;
typedef __attribute__((ext_vector_type(4)))  unsigned int uvec4;

// AS-qualified pointers to int4 chunks (attribute binds to pointee via typedef)
typedef ivec4 __attribute__((address_space(1)))* v4i_gp;
typedef ivec4 __attribute__((address_space(3)))* v4i_lp;

static __device__ inline v4i_gp to_glb_v4(const void* p) {
    return (v4i_gp)(unsigned long long)(uintptr_t)p;
}
static __device__ inline v4i_lp to_lds_v4(const void* p) {
    return (v4i_lp)(unsigned)(uintptr_t)p;     // low 32 bits = LDS byte offset
}

#if __has_builtin(__builtin_amdgcn_global_load_async_to_lds_b128)
#define HAVE_ASYNC_LDS 1
#else
#define HAVE_ASYNC_LDS 0
#endif
#if __has_builtin(__builtin_amdgcn_tensor_load_to_lds)
#define HAVE_TDM 1
#else
#define HAVE_TDM 0
#endif

static __device__ inline void wait_async0() {
#if __has_builtin(__builtin_amdgcn_s_wait_asynccnt)
    __builtin_amdgcn_s_wait_asynccnt(0);
#else
    asm volatile("s_wait_asynccnt 0x0" ::: "memory");
#endif
}

// ---------------------------------------------------------------------------
// helpers
// ---------------------------------------------------------------------------
static __device__ inline __bf16 f2bf(float f) {
    unsigned int u = __builtin_bit_cast(unsigned int, f);
    u += 0x7FFFu + ((u >> 16) & 1u);           // round-to-nearest-even
    unsigned short h = (unsigned short)(u >> 16);
    return __builtin_bit_cast(__bf16, h);
}

static __device__ inline v8f zero_v8f() {
    v8f z;
#pragma unroll
    for (int i = 0; i < 8; ++i) z[i] = 0.0f;
    return z;
}

// Fragment for v_wmma_f32_16x16x32_bf16: lane l owns row/col (l&15); its 16
// K-values are kb..kb+7 and kb+16..kb+23 with kb = (l&16) ? 8 : 0.
static __device__ inline v16bf ld_frag_bf(const __bf16* p, int kb) {
    v16bf f;
#pragma unroll
    for (int i = 0; i < 8; ++i) f[i] = p[kb + i];
#pragma unroll
    for (int i = 0; i < 8; ++i) f[8 + i] = p[kb + 16 + i];
    return f;
}

#define WMMA_BF16(A, B, Cacc) \
    __builtin_amdgcn_wmma_f32_16x16x32_bf16(false, (A), false, (B), (short)0, (Cacc), false, false)

// ---------------------------------------------------------------------------
// Kernel 1: temporal attention.  One workgroup (128 thr = 4 waves) per (b,h,j).
// x tile (120x64 f32, row stride 1536) staged by the Tensor Data Mover.
// ---------------------------------------------------------------------------
#define TSM_XS (128 * 64)   // bf16 x tile (rows >=120 zero)
#define TSM_QS (128 * 32)   // bf16 q (cols 16..31 zero, pre-scaled)
#define TSM_KS (128 * 32)   // bf16 k (cols 16..31 zero)
#define TSM_VT (16 * 128)   // bf16 v transposed VT[a][s]
#define TSM_W  (16 * 64)    // bf16 weights
#define TSM_HALVES (TSM_XS + TSM_QS + TSM_KS + TSM_VT + 3 * TSM_W)
// + P[128][129] f32 + XF[120][64] f32 (TDM landing zone)
#define TSM_BYTES ((size_t)TSM_HALVES * 2 + (size_t)128 * 129 * 4 + (size_t)SS * EE * 4)

__global__ __launch_bounds__(128)
void temporal_attn_kernel(const float* __restrict__ src,
                          const float* __restrict__ tqw,
                          const float* __restrict__ tkw,
                          const float* __restrict__ tvw,
                          float* __restrict__ out) {
    extern __shared__ char smem[];
    __bf16* XS = (__bf16*)smem;
    __bf16* QS = XS + TSM_XS;
    __bf16* KS = QS + TSM_QS;
    __bf16* VT = KS + TSM_KS;
    __bf16* WQ = VT + TSM_VT;
    __bf16* WK = WQ + TSM_W;
    __bf16* WV = WK + TSM_W;
    float*  P  = (float*)(WV + TSM_W);      // [128][129]
    float*  XF = P + 128 * 129;             // [120][64] raw f32 tile

    const int tid  = threadIdx.x;
    const int lane = tid & 31;
    const int wid  = tid >> 5;
    const int nn   = lane & 15;
    const int kb   = (lane & 16) ? 8 : 0;
    const int mhi  = (lane & 16) ? 8 : 0;

    const int bid = blockIdx.x;             // b*H*J + h*J + j
    const int j   = bid % JJ;
    const int h   = (bid / JJ) % HH;
    const int b   = bid / (JJ * HH);

#if HAVE_TDM
    // ---- TDM: DMA the 120x64 f32 tile (row stride 1536 f32) into XF ----
    if (tid < 32) {
        typedef unsigned int u32;
        const unsigned long long ga =
            (unsigned long long)(uintptr_t)(src + (size_t)(b * SS) * CC + j * EE);
        const unsigned lds_off = (unsigned)(uintptr_t)XF;
        uvec4 g0;
        g0[0] = 1u;                                       // count=1, user mode
        g0[1] = lds_off;                                  // lds_addr
        g0[2] = (u32)(ga & 0xFFFFFFFFull);                // global_addr[31:0]
        g0[3] = (u32)((ga >> 32) & 0x1FFFFFFull) | 0x80000000u; // addr[56:32] | type=2
        const unsigned long long st0 = (unsigned long long)CC;  // tensor_dim0_stride
        ivec8 g1;
        g1[0] = (int)(2u << 16);                          // data_size = 4 bytes
        g1[1] = (int)(((u32)EE & 0xFFFFu) << 16);         // tensor_dim0[15:0]
        g1[2] = (int)((((u32)EE >> 16) & 0xFFFFu) | ((u32)SS << 16)); // dim0 hi | dim1 lo
        g1[3] = (int)((((u32)SS >> 16) & 0xFFFFu) | ((u32)EE << 16)); // dim1 hi | tile_dim0
        g1[4] = (int)((u32)SS);                           // tile_dim1=120, tile_dim2=0
        g1[5] = (int)(u32)(st0 & 0xFFFFFFFFull);          // stride0[31:0]
        g1[6] = (int)(u32)((st0 >> 32) & 0xFFFFu);        // stride0[47:32] | stride1 lo=0
        g1[7] = 0;
        ivec4 z4; z4[0] = z4[1] = z4[2] = z4[3] = 0;
        ivec8 z8;
#pragma unroll
        for (int i = 0; i < 8; ++i) z8[i] = 0;
        __builtin_amdgcn_tensor_load_to_lds(g0, g1, z4, z4, z8, 0);
        __builtin_amdgcn_s_wait_tensorcnt(0);
    }
    __syncthreads();
    if (tid < SS) {
        const float* fp = XF + tid * EE;
#pragma unroll
        for (int e = 0; e < EE; e += 4) {
            float4 f = *(const float4*)(fp + e);
            XS[tid * 64 + e + 0] = f2bf(f.x);
            XS[tid * 64 + e + 1] = f2bf(f.y);
            XS[tid * 64 + e + 2] = f2bf(f.z);
            XS[tid * 64 + e + 3] = f2bf(f.w);
        }
    } else {
        for (int e = 0; e < EE; ++e) XS[tid * 64 + e] = f2bf(0.0f);
    }
#else
    if (tid < SS) {
        const float* gp = src + ((size_t)(b * SS + tid) * CC) + j * EE;
#pragma unroll
        for (int e = 0; e < EE; e += 4) {
            float4 f = *(const float4*)(gp + e);
            XS[tid * 64 + e + 0] = f2bf(f.x);
            XS[tid * 64 + e + 1] = f2bf(f.y);
            XS[tid * 64 + e + 2] = f2bf(f.z);
            XS[tid * 64 + e + 3] = f2bf(f.w);
        }
    } else {
        for (int e = 0; e < EE; ++e) XS[tid * 64 + e] = f2bf(0.0f);
    }
#endif

    // ---- stage weights (each 16x64 = 1024 floats) ----
    {
        const size_t wo = (size_t)(h * JJ + j) * (AA * EE);
        for (int i = tid; i < AA * EE; i += 128) {
            WQ[i] = f2bf(tqw[wo + i]);
            WK[i] = f2bf(tkw[wo + i]);
            WV[i] = f2bf(tvw[wo + i]);
        }
    }
    for (int c = 16; c < 32; ++c) {         // zero pad K 16..31 of q/k
        QS[tid * 32 + c] = f2bf(0.0f);
        KS[tid * 32 + c] = f2bf(0.0f);
    }
    __syncthreads();

    // ---- QKV: M=128 (8 tiles of 16), N=16, K=64 (2 WMMA steps) ----
#pragma unroll
    for (int t = 0; t < 2; ++t) {
        const int mt = wid * 2 + t;
        v8f aq = zero_v8f(), ak = zero_v8f(), av = zero_v8f();
#pragma unroll
        for (int k0 = 0; k0 < EE; k0 += 32) {
            v16bf ax = ld_frag_bf(XS + (mt * 16 + nn) * 64 + k0, kb);
            v16bf bq = ld_frag_bf(WQ + nn * 64 + k0, kb);
            v16bf bk = ld_frag_bf(WK + nn * 64 + k0, kb);
            v16bf bv = ld_frag_bf(WV + nn * 64 + k0, kb);
            aq = WMMA_BF16(ax, bq, aq);
            ak = WMMA_BF16(ax, bk, ak);
            av = WMMA_BF16(ax, bv, av);
        }
#pragma unroll
        for (int r = 0; r < 8; ++r) {
            const int m = mt * 16 + r + mhi;
            QS[m * 32 + nn] = f2bf(aq[r] * 0.25f);  // fold in 1/sqrt(A)
            KS[m * 32 + nn] = f2bf(ak[r]);
            VT[nn * 128 + m] = f2bf(av[r]);
        }
    }
    __syncthreads();

    // ---- scores: S = q @ k^T, M=N=128, K=32 (padded from 16) ----
#pragma unroll
    for (int t = 0; t < 2; ++t) {
        const int mt = wid * 2 + t;
        v16bf aq = ld_frag_bf(QS + (mt * 16 + nn) * 32, kb);
#pragma unroll
        for (int nt = 0; nt < 8; ++nt) {
            v16bf bk = ld_frag_bf(KS + (nt * 16 + nn) * 32, kb);
            v8f s = zero_v8f();
            s = WMMA_BF16(aq, bk, s);
#pragma unroll
            for (int r = 0; r < 8; ++r)
                P[(mt * 16 + r + mhi) * 129 + nt * 16 + nn] = s[r];
        }
    }
    __syncthreads();

    // ---- causal softmax, one row per thread ----
    {
        const int r = tid;
        float* row = P + r * 129;
        if (r < SS) {
            float mx = -3.4e38f;
            for (int c = 0; c <= r; ++c) mx = fmaxf(mx, row[c]);
            float sum = 0.0f;
            for (int c = 0; c <= r; ++c) {
                float e = __expf(row[c] - mx);
                row[c] = e; sum += e;
            }
            const float inv = 1.0f / sum;
            for (int c = 0; c <= r; ++c) row[c] *= inv;
            for (int c = r + 1; c < 128; ++c) row[c] = 0.0f;
        } else {
            for (int c = 0; c < 128; ++c) row[c] = 0.0f;
        }
    }
    __syncthreads();

    // ---- AV: t = P @ v, M=128, N=16, K=128 (4 WMMA steps) ----
#pragma unroll
    for (int t = 0; t < 2; ++t) {
        const int mt = wid * 2 + t;
        v8f o = zero_v8f();
#pragma unroll
        for (int k0 = 0; k0 < 128; k0 += 32) {
            const float* pr = P + (mt * 16 + nn) * 129 + k0;
            v16bf pa;
#pragma unroll
            for (int i = 0; i < 8; ++i) pa[i] = f2bf(pr[kb + i]);
#pragma unroll
            for (int i = 0; i < 8; ++i) pa[8 + i] = f2bf(pr[kb + 16 + i]);
            v16bf bv = ld_frag_bf(VT + nn * 128 + k0, kb);
            o = WMMA_BF16(pa, bv, o);
        }
#pragma unroll
        for (int r = 0; r < 8; ++r) {
            const int s = mt * 16 + r + mhi;
            if (s < SS)
                out[((size_t)(b * SS + s)) * CC + h * (JJ * AA) + j * AA + nn] = o[r];
        }
    }
}

// ---------------------------------------------------------------------------
// Kernel 2: bf16 WMMA GEMM  Y = A @ B^T + bias (+ residual)
//   A:[M,K] bf16, B:[N,K] bf16 (pre-converted), Y:[M,N] f32
//   256 thr = 8 waves, tile 128x128, K-step 32; tiles staged with
//   GLOBAL_LOAD_ASYNC_TO_LDS_B128 when available.
// ---------------------------------------------------------------------------
__global__ __launch_bounds__(256)
void gemm_bf16_kernel(const __bf16* __restrict__ A, const __bf16* __restrict__ Bw,
                      const float* __restrict__ bias, const float* __restrict__ res,
                      float* __restrict__ out, int M, int N, int K) {
    __shared__ __bf16 As[128 * 32];
    __shared__ __bf16 Bs[128 * 32];

    const int tid  = threadIdx.x;
    const int lane = tid & 31;
    const int wid  = tid >> 5;
    const int wm   = wid & 3;            // 4 wave-rows of 32
    const int wn   = wid >> 2;           // 2 wave-cols of 64
    const int nn   = lane & 15;
    const int kb   = (lane & 16) ? 8 : 0;
    const int mhi  = (lane & 16) ? 8 : 0;

    const int bM = blockIdx.x * 128;
    const int bN = blockIdx.y * 128;

    v8f acc[2][4];
#pragma unroll
    for (int i = 0; i < 2; ++i)
#pragma unroll
        for (int jn = 0; jn < 4; ++jn) acc[i][jn] = zero_v8f();

    // tile = 128 rows x 32 bf16 = 512 chunks of 16B; thread handles chunks
    // tid and tid+256 for both A and B.
    const int r0 = tid >> 2,         ko0 = (tid & 3) * 8;
    const int r1 = (tid + 256) >> 2, ko1 = ((tid + 256) & 3) * 8;

    for (int k0 = 0; k0 < K; k0 += 32) {
#if HAVE_ASYNC_LDS
        __builtin_amdgcn_global_load_async_to_lds_b128(
            to_glb_v4(A + (size_t)(bM + r0) * K + k0 + ko0), to_lds_v4(As + r0 * 32 + ko0), 0, 0);
        __builtin_amdgcn_global_load_async_to_lds_b128(
            to_glb_v4(A + (size_t)(bM + r1) * K + k0 + ko1), to_lds_v4(As + r1 * 32 + ko1), 0, 0);
        __builtin_amdgcn_global_load_async_to_lds_b128(
            to_glb_v4(Bw + (size_t)(bN + r0) * K + k0 + ko0), to_lds_v4(Bs + r0 * 32 + ko0), 0, 0);
        __builtin_amdgcn_global_load_async_to_lds_b128(
            to_glb_v4(Bw + (size_t)(bN + r1) * K + k0 + ko1), to_lds_v4(Bs + r1 * 32 + ko1), 0, 0);
        wait_async0();
#else
        *(uint4*)(As + r0 * 32 + ko0) = *(const uint4*)(A + (size_t)(bM + r0) * K + k0 + ko0);
        *(uint4*)(As + r1 * 32 + ko1) = *(const uint4*)(A + (size_t)(bM + r1) * K + k0 + ko1);
        *(uint4*)(Bs + r0 * 32 + ko0) = *(const uint4*)(Bw + (size_t)(bN + r0) * K + k0 + ko0);
        *(uint4*)(Bs + r1 * 32 + ko1) = *(const uint4*)(Bw + (size_t)(bN + r1) * K + k0 + ko1);
#endif
        __syncthreads();

        v16bf afr[2], bfr[4];
#pragma unroll
        for (int i = 0; i < 2; ++i)
            afr[i] = ld_frag_bf(As + (wm * 32 + i * 16 + nn) * 32, kb);
#pragma unroll
        for (int jn = 0; jn < 4; ++jn)
            bfr[jn] = ld_frag_bf(Bs + (wn * 64 + jn * 16 + nn) * 32, kb);
#pragma unroll
        for (int i = 0; i < 2; ++i)
#pragma unroll
            for (int jn = 0; jn < 4; ++jn)
                acc[i][jn] = WMMA_BF16(afr[i], bfr[jn], acc[i][jn]);
        __syncthreads();
    }

#pragma unroll
    for (int i = 0; i < 2; ++i) {
#pragma unroll
        for (int jn = 0; jn < 4; ++jn) {
            const int n = bN + wn * 64 + jn * 16 + nn;
            const float bv = bias[n];
#pragma unroll
            for (int r = 0; r < 8; ++r) {
                const int m = bM + wm * 32 + i * 16 + r + mhi;
                const size_t idx = (size_t)m * N + n;
                float v = acc[i][jn][r] + bv;
                if (res) v += res[idx];
                out[idx] = v;
            }
        }
    }
}

// ---------------------------------------------------------------------------
// Kernel 3: spatial attention (J=24 -> VALU path).  One WG per (b,s).
// ---------------------------------------------------------------------------
__global__ __launch_bounds__(128)
void spatial_attn_kernel(const float* __restrict__ src,
                         const float* __restrict__ sqw,
                         const float* __restrict__ skw,
                         const float* __restrict__ svw,
                         float* __restrict__ out) {
    __shared__ float xs[JJ * EE];
    __shared__ float qL[HH * JJ * AA];
    __shared__ float kL[HH * JJ * AA];
    __shared__ float vL[HH * JJ * AA];
    __shared__ float sc[HH * JJ * JJ];

    const int bs  = blockIdx.x;
    const int tid = threadIdx.x;
    const float* xp = src + (size_t)bs * CC;

    for (int i = tid; i < JJ * EE; i += 128) xs[i] = xp[i];
    __syncthreads();

    for (int i = tid; i < HH * JJ * AA; i += 128) {
        const int a = i & (AA - 1);
        const int j = (i >> 4) % JJ;
        const int h = i / (JJ * AA);
        const float* xr = xs + j * EE;
        const float* wq = sqw + ((size_t)(h * JJ + j) * AA + a) * EE;
        const float* wk = skw + ((size_t)h * AA + a) * EE;
        const float* wv = svw + ((size_t)h * AA + a) * EE;
        float sq = 0.f, sk = 0.f, sv = 0.f;
#pragma unroll 4
        for (int e = 0; e < EE; ++e) {
            const float x = xr[e];
            sq += x * wq[e]; sk += x * wk[e]; sv += x * wv[e];
        }
        qL[i] = sq; kL[i] = sk; vL[i] = sv;
    }
    __syncthreads();

    for (int i = tid; i < HH * JJ * JJ; i += 128) {
        const int jk = i % JJ;
        const int jq = (i / JJ) % JJ;
        const int h  = i / (JJ * JJ);
        const float* qr = qL + (h * JJ + jq) * AA;
        const float* kr = kL + (h * JJ + jk) * AA;
        float s = 0.f;
#pragma unroll
        for (int a = 0; a < AA; ++a) s += qr[a] * kr[a];
        sc[i] = s * 0.25f;
    }
    __syncthreads();

    if (tid < HH * JJ) {
        float* row = sc + tid * JJ;
        float mx = -3.4e38f;
        for (int c = 0; c < JJ; ++c) mx = fmaxf(mx, row[c]);
        float sum = 0.f;
        for (int c = 0; c < JJ; ++c) { float e = __expf(row[c] - mx); row[c] = e; sum += e; }
        const float inv = 1.0f / sum;
        for (int c = 0; c < JJ; ++c) row[c] *= inv;
    }
    __syncthreads();

    for (int i = tid; i < HH * JJ * AA; i += 128) {
        const int a = i & (AA - 1);
        const int j = (i >> 4) % JJ;
        const int h = i / (JJ * AA);
        const float* pr = sc + (h * JJ + j) * JJ;
        float s = 0.f;
#pragma unroll
        for (int jk = 0; jk < JJ; ++jk) s += pr[jk] * vL[(h * JJ + jk) * AA + a];
        out[(size_t)bs * CC + h * (JJ * AA) + j * AA + a] = s;
    }
}

// ---------------------------------------------------------------------------
// BatchNorm + elementwise utilities
// ---------------------------------------------------------------------------
__global__ __launch_bounds__(256)
void bn_stats_kernel(const float* __restrict__ x, float* __restrict__ stats,
                     int nrows, int rowsPer) {
    const int ch = blockIdx.x * blockDim.x + threadIdx.x;
    const int r0 = blockIdx.y * rowsPer;
    int r1 = r0 + rowsPer; if (r1 > nrows) r1 = nrows;
    float s = 0.f, s2 = 0.f;
    for (int r = r0; r < r1; ++r) {
        const float v = x[(size_t)r * CC + ch];
        s += v; s2 += v * v;
    }
    atomicAdd(&stats[ch], s);
    atomicAdd(&stats[CC + ch], s2);
}

__global__ __launch_bounds__(256)
void bn_apply_kernel(const float* __restrict__ x, const float* __restrict__ stats,
                     const float* __restrict__ g, const float* __restrict__ bta,
                     float* __restrict__ y, int total) {
    const float invN = 1.0f / (float)NROW;
    for (int i = blockIdx.x * blockDim.x + threadIdx.x; i < total;
         i += gridDim.x * blockDim.x) {
        const int ch = i % CC;
        const float m   = stats[ch] * invN;
        const float var = stats[CC + ch] * invN - m * m;
        y[i] = (x[i] - m) * rsqrtf(var + EPSV) * g[ch] + bta[ch];
    }
}

__global__ __launch_bounds__(256)
void add2_kernel(const float* __restrict__ a, const float* __restrict__ b,
                 float* __restrict__ c, int total) {
    for (int i = blockIdx.x * blockDim.x + threadIdx.x; i < total;
         i += gridDim.x * blockDim.x)
        c[i] = a[i] + b[i];
}

__global__ __launch_bounds__(256)
void cvt_bf16_kernel(const float* __restrict__ x, __bf16* __restrict__ y, int n4) {
    for (int i = blockIdx.x * blockDim.x + threadIdx.x; i < n4;
         i += gridDim.x * blockDim.x) {
        const float4 f = *(const float4*)(x + 4 * (size_t)i);
        __bf16* yp = y + 4 * (size_t)i;
        yp[0] = f2bf(f.x); yp[1] = f2bf(f.y); yp[2] = f2bf(f.z); yp[3] = f2bf(f.w);
    }
}

// ---------------------------------------------------------------------------
// Launcher
// ---------------------------------------------------------------------------
extern "C" void kernel_launch(void* const* d_in, const int* in_sizes, int n_in,
                              void* d_out, int out_size, void* d_ws, size_t ws_size,
                              hipStream_t stream) {
    (void)in_sizes; (void)n_in; (void)out_size; (void)ws_size;

    const float* src  = (const float*)d_in[0];
    const float* tq_w = (const float*)d_in[1];
    const float* tk_w = (const float*)d_in[2];
    const float* tv_w = (const float*)d_in[3];
    const float* t_pw = (const float*)d_in[4];
    const float* t_pb = (const float*)d_in[5];
    const float* t_g  = (const float*)d_in[6];
    const float* t_b  = (const float*)d_in[7];
    const float* sq_w = (const float*)d_in[8];
    const float* sk_w = (const float*)d_in[9];
    const float* sv_w = (const float*)d_in[10];
    const float* s_pw = (const float*)d_in[11];
    const float* s_pb = (const float*)d_in[12];
    const float* s_g  = (const float*)d_in[13];
    const float* s_b  = (const float*)d_in[14];
    const float* fc_w = (const float*)d_in[15];
    const float* fc_b = (const float*)d_in[16];
    const float* fc_g = (const float*)d_in[17];
    const float* fc_bb= (const float*)d_in[18];
    float* outp = (float*)d_out;

    const size_t NTOT = (size_t)NROW * CC;
    const size_t WTOT = (size_t)CC * CC;
    float*  bufA  = (float*)d_ws;        // attention outputs / t+sp sum
    float*  bufB  = bufA + NTOT;         // temporal branch
    float*  bufC  = bufB + NTOT;         // spatial branch / fc output
    float*  stats = bufC + NTOT;         // 2*CC
    __bf16* Abf   = (__bf16*)(stats + 2 * CC);  // bf16 activations
    __bf16* Wbf   = Abf + NTOT;                 // bf16 weights

    const dim3 gemmGrid(NROW / 128, CC / 128);
    const dim3 statGrid(CC / 256, 32);
    const int rowsPer = (NROW + 31) / 32;
    const int ewGrid = 4096;

    // ---- temporal branch ----
    temporal_attn_kernel<<<BB * HH * JJ, 128, TSM_BYTES, stream>>>(
        src, tq_w, tk_w, tv_w, bufA);
    cvt_bf16_kernel<<<ewGrid, 256, 0, stream>>>(bufA, Abf, (int)(NTOT / 4));
    cvt_bf16_kernel<<<ewGrid, 256, 0, stream>>>(t_pw, Wbf, (int)(WTOT / 4));
    gemm_bf16_kernel<<<gemmGrid, 256, 0, stream>>>(
        Abf, Wbf, t_pb, src, bufB, NROW, CC, CC);
    (void)hipMemsetAsync(stats, 0, 2 * CC * sizeof(float), stream);
    bn_stats_kernel<<<statGrid, 256, 0, stream>>>(bufB, stats, NROW, rowsPer);
    bn_apply_kernel<<<ewGrid, 256, 0, stream>>>(bufB, stats, t_g, t_b, bufB, (int)NTOT);

    // ---- spatial branch ----
    spatial_attn_kernel<<<NROW, 128, 0, stream>>>(src, sq_w, sk_w, sv_w, bufA);
    cvt_bf16_kernel<<<ewGrid, 256, 0, stream>>>(bufA, Abf, (int)(NTOT / 4));
    cvt_bf16_kernel<<<ewGrid, 256, 0, stream>>>(s_pw, Wbf, (int)(WTOT / 4));
    gemm_bf16_kernel<<<gemmGrid, 256, 0, stream>>>(
        Abf, Wbf, s_pb, src, bufC, NROW, CC, CC);
    (void)hipMemsetAsync(stats, 0, 2 * CC * sizeof(float), stream);
    bn_stats_kernel<<<statGrid, 256, 0, stream>>>(bufC, stats, NROW, rowsPer);
    bn_apply_kernel<<<ewGrid, 256, 0, stream>>>(bufC, stats, s_g, s_b, bufC, (int)NTOT);

    // ---- combine + fc + final BN ----
    add2_kernel<<<ewGrid, 256, 0, stream>>>(bufB, bufC, bufA, (int)NTOT);
    cvt_bf16_kernel<<<ewGrid, 256, 0, stream>>>(bufA, Abf, (int)(NTOT / 4));
    cvt_bf16_kernel<<<ewGrid, 256, 0, stream>>>(fc_w, Wbf, (int)(WTOT / 4));
    gemm_bf16_kernel<<<gemmGrid, 256, 0, stream>>>(
        Abf, Wbf, fc_b, bufA, bufC, NROW, CC, CC);
    (void)hipMemsetAsync(stats, 0, 2 * CC * sizeof(float), stream);
    bn_stats_kernel<<<statGrid, 256, 0, stream>>>(bufC, stats, NROW, rowsPer);
    bn_apply_kernel<<<ewGrid, 256, 0, stream>>>(bufC, stats, fc_g, fc_bb, outp, (int)NTOT);
}
